// YICAOptimizedAttention_44332652429325
// MI455X (gfx1250) — matrix operand order repaired and checked by reading, not compiled, not run on hardware
//
#include <hip/hip_runtime.h>

typedef __attribute__((ext_vector_type(16))) __bf16 v16bf;
typedef __attribute__((ext_vector_type(8)))  __bf16 v8bf;
typedef __attribute__((ext_vector_type(8)))  float  v8f;
typedef __attribute__((ext_vector_type(4)))  int    v4i;

#define AS_GLOBAL __attribute__((address_space(1)))
#define AS_LDS    __attribute__((address_space(3)))

#if defined(__has_builtin)
#  if __has_builtin(__builtin_amdgcn_global_load_async_to_lds_b128)
#    define HAVE_ASYNC_LDS 1
#  endif
#endif
#ifndef HAVE_ASYNC_LDS
#  define HAVE_ASYNC_LDS 0
#endif

#if HAVE_ASYNC_LDS
#  if __has_builtin(__builtin_amdgcn_s_wait_asynccnt)
#    define WAIT_ASYNC() __builtin_amdgcn_s_wait_asynccnt(0)
#  else
#    define WAIT_ASYNC() asm volatile("s_wait_asynccnt 0x0" ::: "memory")
#  endif
#else
#  define WAIT_ASYNC() ((void)0)
#endif

__device__ __forceinline__ v16bf combo16(v8bf lo, v8bf hi) {
  union { v16bf v; v8bf h[2]; } u;
  u.h[0] = lo; u.h[1] = hi;
  return u.v;
}
__device__ __forceinline__ v8f vzero8() {
  v8f z;
#pragma unroll
  for (int i = 0; i < 8; ++i) z[i] = 0.0f;
  return z;
}

// ---------------------------------------------------------------------------
// Generic NT GEMM: C[M,N] = alpha * (A[M,K] @ B[N,K]^T)
// A: fp32 or bf16 row-major (lda=K); B: fp32 row-major (ldb=K); C: fp32/bf16.
// Block 256 thr (8 waves), tile 128x128, BK=32, bf16 WMMA w/ f32 accumulate.
// ---------------------------------------------------------------------------
#define BM 128
#define BN 128
#define BK 32
#define LDT 40   // padded LDS row (bf16 elems): 80B = 20 banks, 16B aligned

template <typename TA, typename TO>
__global__ __launch_bounds__(256) void gemm_nt_kernel(
    const TA* __restrict__ A, const float* __restrict__ Bw, TO* __restrict__ C,
    int M, int N, int K, float alpha)
{
  __shared__ alignas(16) __bf16 As[2][BM][LDT];
  __shared__ alignas(16) __bf16 Bs[2][BN][LDT];

  const int tid  = threadIdx.x;
  const int wave = tid >> 5;
  const int lane = tid & 31;
  const int lm   = lane & 15;
  const int hf   = lane >> 4;

  const int m0 = blockIdx.y * BM;
  const int n0 = blockIdx.x * BN;
  const int wm = (wave & 1) * 64;   // 2 waves along M
  const int wn = (wave >> 1) * 32;  // 4 waves along N

  v8f acc[4][2];
#pragma unroll
  for (int i = 0; i < 4; ++i)
#pragma unroll
    for (int j = 0; j < 2; ++j) acc[i][j] = vzero8();

  auto loadA = [&](int kt, int st) {
    if constexpr (sizeof(TA) == 4) {
#pragma unroll
      for (int i = 0; i < 4; ++i) {
        const int r = i * 32 + (tid >> 3);
        const int c = (tid & 7) * 4;
        const float4 v = *reinterpret_cast<const float4*>(
            A + (size_t)(m0 + r) * K + kt + c);
        union { __bf16 h[4]; unsigned long long u; } pk;
        pk.h[0] = (__bf16)v.x; pk.h[1] = (__bf16)v.y;
        pk.h[2] = (__bf16)v.z; pk.h[3] = (__bf16)v.w;
        *reinterpret_cast<unsigned long long*>(&As[st][r][c]) = pk.u;
      }
    } else {
#pragma unroll
      for (int i = 0; i < 2; ++i) {
        const int idx = tid + i * 256;
        const int r = idx >> 2;
        const int c = (idx & 3) * 8;
        *reinterpret_cast<uint4*>(&As[st][r][c]) =
            *reinterpret_cast<const uint4*>(A + (size_t)(m0 + r) * K + kt + c);
      }
    }
  };
  auto loadB = [&](int kt, int st) {
#pragma unroll
    for (int i = 0; i < 4; ++i) {
      const int r = i * 32 + (tid >> 3);
      const int c = (tid & 7) * 4;
      const float4 v = *reinterpret_cast<const float4*>(
          Bw + (size_t)(n0 + r) * K + kt + c);
      union { __bf16 h[4]; unsigned long long u; } pk;
      pk.h[0] = (__bf16)v.x; pk.h[1] = (__bf16)v.y;
      pk.h[2] = (__bf16)v.z; pk.h[3] = (__bf16)v.w;
      *reinterpret_cast<unsigned long long*>(&Bs[st][r][c]) = pk.u;
    }
  };

  loadA(0, 0); loadB(0, 0);
  __syncthreads();
  const int NK = K / BK;
  for (int kt = 0; kt < NK; ++kt) {
    const int st = kt & 1;
    if (kt + 1 < NK) { loadA((kt + 1) * BK, st ^ 1); loadB((kt + 1) * BK, st ^ 1); }
    if (kt + 2 < NK) {  // gfx1250 global_prefetch_b8 hints, 2 tiles ahead
      const int kp = (kt + 2) * BK;
      __builtin_prefetch(A  + (size_t)(m0 + (tid >> 1)) * K + kp + (tid & 1) * 16, 0, 1);
      __builtin_prefetch(Bw + (size_t)(n0 + (tid >> 1)) * K + kp + (tid & 1) * 16, 0, 1);
    }

    v16bf af[4], bfr[2];
#pragma unroll
    for (int mi = 0; mi < 4; ++mi) {
      const __bf16* p = &As[st][wm + mi * 16 + lm][0];
      af[mi] = combo16(*(const v8bf*)(p + 8 * hf),
                       *(const v8bf*)(p + 16 + 8 * hf));
    }
#pragma unroll
    for (int ni = 0; ni < 2; ++ni) {
      const __bf16* p = &Bs[st][wn + ni * 16 + lm][0];
      bfr[ni] = combo16(*(const v8bf*)(p + 8 * hf),
                        *(const v8bf*)(p + 16 + 8 * hf));
    }
#pragma unroll
    for (int mi = 0; mi < 4; ++mi)
#pragma unroll
      for (int ni = 0; ni < 2; ++ni)
        acc[mi][ni] = __builtin_amdgcn_wmma_f32_16x16x32_bf16(
            false, af[mi], false, bfr[ni], (short)0, acc[mi][ni], false, false);
    __syncthreads();
  }

#pragma unroll
  for (int mi = 0; mi < 4; ++mi)
#pragma unroll
    for (int ni = 0; ni < 2; ++ni)
#pragma unroll
      for (int r = 0; r < 8; ++r) {
        const int row = m0 + wm + mi * 16 + hf * 8 + r;
        const int col = n0 + wn + ni * 16 + lm;
        C[(size_t)row * N + col] = (TO)(acc[mi][ni][r] * alpha);
      }
}

// ---------------------------------------------------------------------------
// Flash attention: Q,K,V bf16 in [b,s,h,hd] layout; Q pre-scaled by 1/sqrt(hd).
// One block = (b, h, 128 queries); each wave owns 16 queries x HD=128.
// Streams 32-key chunks through double-buffered LDS; online softmax in f32.
// ---------------------------------------------------------------------------
#define FS  2048
#define FH  2048
#define FNH 16
#define FHD 128
#define KC  32
#define KPAD 136  // 272B rows (17*16B) -> conflict-free b128 reads
#define VPAD 40

__global__ __launch_bounds__(256) void flash_attn_kernel(
    const __bf16* __restrict__ Q, const __bf16* __restrict__ Kb,
    const __bf16* __restrict__ Vb, const float* __restrict__ mask,
    __bf16* __restrict__ O)
{
  __shared__ alignas(16) __bf16 Ks[2][KC][KPAD];     // [key][hd]
  __shared__ alignas(16) __bf16 Vts[2][FHD][VPAD];   // transposed [hd][key]
  __shared__ alignas(16) __bf16 Ps[8][16][VPAD];     // per-wave P tile [q][key]

  const int tid  = threadIdx.x;
  const int wave = tid >> 5;
  const int lane = tid & 31;
  const int lm   = lane & 15;
  const int hf   = lane >> 4;

  const int b  = blockIdx.z;
  const int h  = blockIdx.y;
  const int q0 = blockIdx.x * 128;
  const int qw = q0 + wave * 16;

  // Q fragments (A-matrix, 4 chunks of K=32 over HD=128), loaded once.
  v16bf qf[4];
  {
    const __bf16* qp = Q + ((size_t)(b * FS + qw + lm) * FH) + h * FHD;
#pragma unroll
    for (int c = 0; c < 4; ++c)
      qf[c] = combo16(*(const v8bf*)(qp + c * 32 + 16 * hf),
                      *(const v8bf*)(qp + c * 32 + 16 * hf + 8));
  }

  v8f acc[8];
#pragma unroll
  for (int i = 0; i < 8; ++i) acc[i] = vzero8();
  float mrow[8], lrow[8];
#pragma unroll
  for (int r = 0; r < 8; ++r) { mrow[r] = -1e30f; lrow[r] = 0.0f; }

  auto loadKV = [&](int c, int st) {
    const int k0 = c * KC;
    // K tile: straight byte copy -> async DMA to LDS when available
#pragma unroll
    for (int i = 0; i < 2; ++i) {
      const int idx = tid + i * 256;
      const int r   = idx >> 4;         // 0..31 key rows
      const int col = (idx & 15) * 8;   // hd chunks of 8
      const __bf16* src = Kb + ((size_t)(b * FS + k0 + r) * FH) + h * FHD + col;
#if HAVE_ASYNC_LDS
      __builtin_amdgcn_global_load_async_to_lds_b128(
          (AS_GLOBAL v4i*)src, (AS_LDS v4i*)&Ks[st][r][col], 0, 0);
#else
      *reinterpret_cast<uint4*>(&Ks[st][r][col]) =
          *reinterpret_cast<const uint4*>(src);
#endif
    }
    // V tile: transpose through registers (needed for B-fragment layout)
#pragma unroll
    for (int i = 0; i < 2; ++i) {
      const int idx = tid + i * 256;
      const int r   = idx >> 4;
      const int col = (idx & 15) * 8;
      union { uint4 u; __bf16 e[8]; } cv;
      cv.u = *reinterpret_cast<const uint4*>(
          Vb + ((size_t)(b * FS + k0 + r) * FH) + h * FHD + col);
#pragma unroll
      for (int j = 0; j < 8; ++j) Vts[st][col + j][r] = cv.e[j];
    }
  };

  loadKV(0, 0);
  WAIT_ASYNC();
  __syncthreads();

  const float* mp = mask + (size_t)b * FS * FS;  // mask[b][0][q][k]
  const int NC = FS / KC;  // 64
  for (int c = 0; c < NC; ++c) {
    const int st = c & 1;
    if (c + 1 < NC) loadKV(c + 1, st ^ 1);
    if (c + 2 < NC) {  // prefetch hints two chunks ahead
      const size_t off = ((size_t)(b * FS + (c + 2) * KC + (tid >> 3)) * FH) +
                         h * FHD + (tid & 7) * 16;
      __builtin_prefetch(Kb + off, 0, 1);
      __builtin_prefetch(Vb + off, 0, 1);
    }
    const int k0 = c * KC;

    // ---- scores: preload all 8 K B-fragments, then burst 8 WMMAs ----
    v16bf kfr[8];
#pragma unroll
    for (int c4 = 0; c4 < 4; ++c4) {
      const __bf16* p0 = &Ks[st][lm][c4 * 32 + 16 * hf];
      const __bf16* p1 = &Ks[st][16 + lm][c4 * 32 + 16 * hf];
      kfr[c4]     = combo16(*(const v8bf*)p0, *(const v8bf*)(p0 + 8));
      kfr[4 + c4] = combo16(*(const v8bf*)p1, *(const v8bf*)(p1 + 8));
    }
    v8f s0 = vzero8(), s1 = vzero8();
#pragma unroll
    for (int c4 = 0; c4 < 4; ++c4) {
      s0 = __builtin_amdgcn_wmma_f32_16x16x32_bf16(false, qf[c4], false, kfr[c4],
                                                   (short)0, s0, false, false);
      s1 = __builtin_amdgcn_wmma_f32_16x16x32_bf16(false, qf[c4], false, kfr[4 + c4],
                                                   (short)0, s1, false, false);
    }

    // ---- mask add + online softmax (row m = hf*8+r lives across 16 lanes) ----
    float alp[8];
#pragma unroll
    for (int r = 0; r < 8; ++r) {
      const int q = qw + hf * 8 + r;
      s0[r] += mp[(size_t)q * FS + k0 + lm];
      s1[r] += mp[(size_t)q * FS + k0 + 16 + lm];
      float mx = fmaxf(s0[r], s1[r]);
      mx = fmaxf(mx, __shfl_xor(mx, 1, 32));
      mx = fmaxf(mx, __shfl_xor(mx, 2, 32));
      mx = fmaxf(mx, __shfl_xor(mx, 4, 32));
      mx = fmaxf(mx, __shfl_xor(mx, 8, 32));
      const float mnew = fmaxf(mrow[r], mx);
      const float a = __expf(mrow[r] - mnew);
      mrow[r] = mnew;
      s0[r] = __expf(s0[r] - mnew);
      s1[r] = __expf(s1[r] - mnew);
      float rs = s0[r] + s1[r];
      rs += __shfl_xor(rs, 1, 32);
      rs += __shfl_xor(rs, 2, 32);
      rs += __shfl_xor(rs, 4, 32);
      rs += __shfl_xor(rs, 8, 32);
      lrow[r] = lrow[r] * a + rs;
      alp[r] = a;
    }
#pragma unroll
    for (int nt = 0; nt < 8; ++nt)
#pragma unroll
      for (int r = 0; r < 8; ++r) acc[nt][r] *= alp[r];

    // ---- re-layout P (C-layout -> A-fragment) through per-wave LDS ----
#pragma unroll
    for (int r = 0; r < 8; ++r) {
      const int qq = hf * 8 + r;
      Ps[wave][qq][lm]      = (__bf16)s0[r];
      Ps[wave][qq][16 + lm] = (__bf16)s1[r];
    }
    asm volatile("s_wait_dscnt 0" ::: "memory");
    const __bf16* pp = &Ps[wave][lm][0];
    v16bf pf = combo16(*(const v8bf*)(pp + 16 * hf),
                       *(const v8bf*)(pp + 16 * hf + 8));

    // ---- O += P @ V: preload all 8 V B-fragments, then burst 8 WMMAs ----
    v16bf vf[8];
#pragma unroll
    for (int nt = 0; nt < 8; ++nt) {
      const __bf16* vp = &Vts[st][nt * 16 + lm][16 * hf];
      vf[nt] = combo16(*(const v8bf*)vp, *(const v8bf*)(vp + 8));
    }
#pragma unroll
    for (int nt = 0; nt < 8; ++nt)
      acc[nt] = __builtin_amdgcn_wmma_f32_16x16x32_bf16(false, pf, false, vf[nt],
                                                        (short)0, acc[nt],
                                                        false, false);
    WAIT_ASYNC();
    __syncthreads();
  }

  // normalize + store as bf16 in [b, s, h, hd] (row-major 4096 x 2048)
  float inv[8];
#pragma unroll
  for (int r = 0; r < 8; ++r) inv[r] = 1.0f / lrow[r];
#pragma unroll
  for (int nt = 0; nt < 8; ++nt)
#pragma unroll
    for (int r = 0; r < 8; ++r) {
      const int q = qw + hf * 8 + r;
      O[((size_t)(b * FS + q) * FH) + h * FHD + nt * 16 + lm] =
          (__bf16)(acc[nt][r] * inv[r]);
    }
}

// ---------------------------------------------------------------------------
extern "C" void kernel_launch(void* const* d_in, const int* in_sizes, int n_in,
                              void* d_out, int out_size, void* d_ws, size_t ws_size,
                              hipStream_t stream) {
  (void)in_sizes; (void)n_in; (void)out_size; (void)ws_size;
  const float* x    = (const float*)d_in[0];
  const float* mask = (const float*)d_in[1];
  const float* wq   = (const float*)d_in[2];
  const float* wk   = (const float*)d_in[3];
  const float* wv   = (const float*)d_in[4];
  const float* wo   = (const float*)d_in[5];
  float* out = (float*)d_out;

  const int Mtot = 2 * FS;  // 4096 rows (b*s)
  const int Hd   = FH;      // 2048

  char* ws = (char*)d_ws;
  const size_t bufBytes = (size_t)Mtot * Hd * sizeof(__bf16);  // 16 MB
  __bf16* Qbf = (__bf16*)(ws);
  __bf16* Kbf = (__bf16*)(ws + bufBytes);
  __bf16* Vbf = (__bf16*)(ws + 2 * bufBytes);
  __bf16* Obf = (__bf16*)(ws + 3 * bufBytes);

  const dim3 gblk(Hd / BN, Mtot / BM);  // (16, 32)
  const float qscale = 0.08838834764831845f;  // 1/sqrt(128)

  gemm_nt_kernel<float, __bf16><<<gblk, 256, 0, stream>>>(x, wq, Qbf, Mtot, Hd, Hd, qscale);
  gemm_nt_kernel<float, __bf16><<<gblk, 256, 0, stream>>>(x, wk, Kbf, Mtot, Hd, Hd, 1.0f);
  gemm_nt_kernel<float, __bf16><<<gblk, 256, 0, stream>>>(x, wv, Vbf, Mtot, Hd, Hd, 1.0f);

  flash_attn_kernel<<<dim3(FS / 128, FNH, 2), 256, 0, stream>>>(Qbf, Kbf, Vbf, mask, Obf);

  gemm_nt_kernel<__bf16, float><<<gblk, 256, 0, stream>>>(Obf, wo, out, Mtot, Hd, Hd, 1.0f);
}
